// MultiheadAttention_27822798144020
// MI455X (gfx1250) — matrix operand order repaired and checked by reading, not compiled, not run on hardware
//
#include <hip/hip_runtime.h>
#include <hip/hip_fp16.h>

typedef _Float16 v16h __attribute__((ext_vector_type(16)));
typedef _Float16 v8h  __attribute__((ext_vector_type(8)));
typedef float    v8f  __attribute__((ext_vector_type(8)));

#define E_DIM   1024
#define H_NUM   16
#define HD      64
#define R_DIM   16
#define T_LEN   2048
#define B_SZ    2
#define S_LEN   2048
#define N_ROWS  (T_LEN * B_SZ)      /* 4096 */
#define LORA_SCALING 1.0f           /* alpha/r = 16/16 */
#define QK_SCALE     0.125f         /* 1/sqrt(64) */
#define SPITCH  2056                /* 2048 + 8 halfs pad */

#define WMMA_F16(A, B, C) __builtin_amdgcn_wmma_f32_16x16x32_f16( \
        false, (A), false, (B), (short)0, (C), false, false)

static __device__ __forceinline__ v16h combine8(v8h lo, v8h hi) {
    v16h r;
#pragma unroll
    for (int i = 0; i < 8; ++i) { r[i] = lo[i]; r[i + 8] = hi[i]; }
    return r;
}

// ---------------------------------------------------------------- prep kernels
__global__ void cvt_f32_to_f16(const float* __restrict__ src,
                               _Float16* __restrict__ dst, int n) {
    int i = blockIdx.x * blockDim.x + threadIdx.x;
    int stride = gridDim.x * blockDim.x;
    for (; i < n; i += stride) dst[i] = (_Float16)src[i];
}

// W_eff[j,k] = W[j,k] + s * sum_r LB[j,r] * LA[r,k]   (then cast f16)
__global__ void build_weff(const float* __restrict__ w,
                           const float* __restrict__ la,
                           const float* __restrict__ lb,
                           _Float16* __restrict__ dst) {
    int idx = blockIdx.x * blockDim.x + threadIdx.x;   // E*E threads
    int j = idx >> 10;
    int k = idx & (E_DIM - 1);
    float acc = w[idx];
#pragma unroll
    for (int r = 0; r < R_DIM; ++r)
        acc += LORA_SCALING * lb[j * R_DIM + r] * la[r * E_DIM + k];
    dst[idx] = (_Float16)acc;
}

__global__ void zero_f32(float* __restrict__ p, int n) {
    int i = blockIdx.x * blockDim.x + threadIdx.x;
    int stride = gridDim.x * blockDim.x;
    for (; i < n; i += stride) p[i] = 0.0f;
}

// ---------------------------------------------------------------- projection GEMM
// Y = X(4096x1024,f16) @ W_eff^T(1024x1024,f16 row-major) + bias, scaled.
// Block = 8 waves; wave -> 32 rows (2 M tiles) x 64 cols (4 N tiles) = 8 WMMA/step.
// Ping-pong double buffering: buffer reloaded only after its math retires, so the
// loop-carried fragments coalesce into fixed registers (no v_mov rotation).
// mode 0: f16 out, head-major  dst[((b*H+h)*SEQ + t)*64 + d]   (q / k)
// mode 1: f16 out, transposed  dst[((b*H+h)*64 + d)*SEQ + t]   (v)
// mode 2: f32 out, row-major   dst[i*E + j]                    (o -> d_out)
struct GFrag {
    v8h a0l, a0h, a1l, a1h;     // two 16-row A fragments (lo/hi K chunks)
    v8h b0[4], b1[4];           // four 16-col B fragments
};

static __device__ __forceinline__ GFrag gload(const _Float16* arow0,
                                              const _Float16* arow1,
                                              const _Float16* const bp[4], int k) {
    GFrag f;
    f.a0l = *(const v8h*)(arow0 + k);
    f.a0h = *(const v8h*)(arow0 + k + 16);
    f.a1l = *(const v8h*)(arow1 + k);
    f.a1h = *(const v8h*)(arow1 + k + 16);
#pragma unroll
    for (int nt = 0; nt < 4; ++nt) {
        f.b0[nt] = *(const v8h*)(bp[nt] + k);
        f.b1[nt] = *(const v8h*)(bp[nt] + k + 16);
    }
    return f;
}

static __device__ __forceinline__ void gmath(const GFrag& f, v8f acc[2][4]) {
    v16h A0 = combine8(f.a0l, f.a0h);
    v16h A1 = combine8(f.a1l, f.a1h);
#pragma unroll
    for (int nt = 0; nt < 4; ++nt) {
        v16h Bt = combine8(f.b0[nt], f.b1[nt]);
        acc[0][nt] = WMMA_F16(A0, Bt, acc[0][nt]);
        acc[1][nt] = WMMA_F16(A1, Bt, acc[1][nt]);
    }
}

__global__ __launch_bounds__(256)
void gemm_proj(const _Float16* __restrict__ X, const _Float16* __restrict__ W,
               const float* __restrict__ bias, void* __restrict__ out,
               int mode, float outscale) {
    const int lane = threadIdx.x & 31;
    const int wave = threadIdx.x >> 5;
    const int lm   = lane & 15;
    const int hi   = lane >> 4;
    const int m_base = blockIdx.x * 256 + wave * 32;
    const int n_base = blockIdx.y * 64;

    const _Float16* arow0 = X + (size_t)(m_base + lm) * E_DIM + (hi ? 8 : 0);
    const _Float16* arow1 = arow0 + 16 * E_DIM;
    const _Float16* bp[4];
#pragma unroll
    for (int nt = 0; nt < 4; ++nt)
        bp[nt] = W + (size_t)(n_base + nt * 16 + lm) * E_DIM + (hi ? 8 : 0);

    v8f acc[2][4] = {};

    GFrag f0 = gload(arow0, arow1, bp, 0);
    for (int k0 = 0; k0 < E_DIM; k0 += 64) {
        GFrag f1 = gload(arow0, arow1, bp, k0 + 32);      // k0+32 <= 992, in range
        __builtin_prefetch(arow0 + k0 + 512, 0, 1);        // global_prefetch_b8
        __builtin_prefetch(bp[0] + k0 + 512, 0, 1);
        gmath(f0, acc);                                    // consumes f0
        const int kn = (k0 + 64 < E_DIM) ? k0 + 64 : 0;    // tail wraps, unused
        f0 = gload(arow0, arow1, bp, kn);                  // refill dead buffer
        gmath(f1, acc);                                    // consumes f1
    }

#pragma unroll
    for (int mt = 0; mt < 2; ++mt) {
#pragma unroll
        for (int nt = 0; nt < 4; ++nt) {
#pragma unroll
            for (int v = 0; v < 8; ++v) {
                int i = m_base + mt * 16 + v + 8 * hi;     // row  (t*B + b)
                int j = n_base + nt * 16 + lm;             // col  (h*64 + d)
                float val = (acc[mt][nt][v] + bias[j]) * outscale;
                if (mode == 2) {
                    ((float*)out)[(size_t)i * E_DIM + j] = val;
                } else {
                    int t = i >> 1, b = i & 1;
                    int h = j >> 6, d = j & 63;
                    _Float16* o = (_Float16*)out;
                    if (mode == 0)
                        o[((size_t)(b * H_NUM + h) * S_LEN + t) * HD + d] = (_Float16)val;
                    else
                        o[((size_t)(b * H_NUM + h) * HD + d) * S_LEN + t] = (_Float16)val;
                }
            }
        }
    }
}

// ---------------------------------------------------------------- fused attention
// Grid: (T/32, B*H). Block = 8 waves, 32 query rows, full S = 2048.
// Phase A: scores = (q*scale) @ k^T via WMMA -> LDS f16 [32][SPITCH]
// Phase B: row softmax (vectorized v8h passes + shfl reductions), += attn_w/H
// Phase C: out = P @ V via WMMA (A from LDS, B from pre-transposed V)
struct KFrag { v8h k0, k1, k2, k3; };
struct PVFrag { v8h p0, p1, u0, u1; };

__global__ __launch_bounds__(256)
void attn_kernel(const _Float16* __restrict__ qh, const _Float16* __restrict__ kh,
                 const _Float16* __restrict__ vT, _Float16* __restrict__ attn_out,
                 float* __restrict__ attn_w) {
    extern __shared__ _Float16 sS[];                   // [32][SPITCH] ~128.5 KB
    const int lane = threadIdx.x & 31;
    const int wave = threadIdx.x >> 5;
    const int lm   = lane & 15;
    const int hi   = lane >> 4;
    const int bh   = blockIdx.y;
    const int b    = bh >> 4;
    const int h    = bh & 15;
    const int t0   = blockIdx.x * 32;
    const int mh   = wave >> 2;                        // m half (0..1)
    const int sq   = wave & 3;                         // s quarter

    // ---- Phase A: scores (ping-pong over 32 s-tiles)
    const _Float16* qrow = qh + ((size_t)bh * T_LEN + t0 + mh * 16 + lm) * HD + (hi ? 8 : 0);
    v16h aq0 = combine8(*(const v8h*)(qrow +  0), *(const v8h*)(qrow + 16));
    v16h aq1 = combine8(*(const v8h*)(qrow + 32), *(const v8h*)(qrow + 48));

    const _Float16* kbase = kh + ((size_t)bh * S_LEN + sq * 512 + lm) * HD + (hi ? 8 : 0);
    auto kload = [&](int st) {
        const _Float16* p = kbase + (size_t)st * (16 * HD);
        KFrag f;
        f.k0 = *(const v8h*)(p +  0);
        f.k1 = *(const v8h*)(p + 16);
        f.k2 = *(const v8h*)(p + 32);
        f.k3 = *(const v8h*)(p + 48);
        return f;
    };
    auto ktile = [&](const KFrag& f, int st) {
        v8f c = {};
        c = WMMA_F16(aq0, combine8(f.k0, f.k1), c);
        c = WMMA_F16(aq1, combine8(f.k2, f.k3), c);
        int s0 = sq * 512 + st * 16;
#pragma unroll
        for (int v = 0; v < 8; ++v)
            sS[(size_t)(mh * 16 + v + 8 * hi) * SPITCH + s0 + lm] = (_Float16)c[v];
    };

    KFrag kf0 = kload(0);
    for (int st = 0; st < 32; st += 2) {
        KFrag kf1 = kload(st + 1);
        ktile(kf0, st);
        kf0 = kload((st + 2 < 32) ? st + 2 : 0);
        ktile(kf1, st + 1);
    }
    __syncthreads();

    // ---- Phase B: softmax; 8 threads per row, each owns a contiguous 256-col chunk
    {
        const int row = threadIdx.x >> 3;
        const int sub = threadIdx.x & 7;
        _Float16* srow = sS + (size_t)row * SPITCH + sub * 256;

        float m = -1e30f;
        for (int s = 0; s < 256; s += 8) {
            v8h x = *(const v8h*)(srow + s);
#pragma unroll
            for (int i = 0; i < 8; ++i) m = fmaxf(m, (float)x[i]);
        }
        m = fmaxf(m, __shfl_xor(m, 1, 8));
        m = fmaxf(m, __shfl_xor(m, 2, 8));
        m = fmaxf(m, __shfl_xor(m, 4, 8));

        float sum = 0.0f;
        for (int s = 0; s < 256; s += 8) {
            v8h x = *(const v8h*)(srow + s);
            v8h e;
#pragma unroll
            for (int i = 0; i < 8; ++i) {
                float p = __expf((float)x[i] - m);
                sum += p;
                e[i] = (_Float16)p;
            }
            *(v8h*)(srow + s) = e;          // store un-normalized exp
        }
        sum += __shfl_xor(sum, 1, 8);
        sum += __shfl_xor(sum, 2, 8);
        sum += __shfl_xor(sum, 4, 8);
        float inv = 1.0f / sum;
        float invH = inv * (1.0f / H_NUM);

        float* wrow = attn_w + ((size_t)b * T_LEN + (t0 + row)) * S_LEN + sub * 256;
        for (int s = 0; s < 256; s += 8) {
            v8h x = *(const v8h*)(srow + s);
            v8h p;
#pragma unroll
            for (int i = 0; i < 8; ++i) {
                float u = (float)x[i];
                p[i] = (_Float16)(u * inv);
                __hip_atomic_fetch_add(&wrow[s + i], u * invH,
                                       __ATOMIC_RELAXED, __HIP_MEMORY_SCOPE_AGENT);
            }
            *(v8h*)(srow + s) = p;          // normalized probabilities, f16
        }
    }
    __syncthreads();

    // ---- Phase C: out = P @ V ; wave -> (mh, d-tile = wave&3), full K = S
    {
        const int nt = wave & 3;
        const _Float16* vrow  = vT + ((size_t)bh * HD + nt * 16 + lm) * S_LEN + (hi ? 8 : 0);
        const _Float16* parow = sS + (size_t)(mh * 16 + lm) * SPITCH + (hi ? 8 : 0);

        auto pvload = [&](int s) {
            PVFrag f;
            f.p0 = *(const v8h*)(parow + s);
            f.p1 = *(const v8h*)(parow + s + 16);
            f.u0 = *(const v8h*)(vrow + s);
            f.u1 = *(const v8h*)(vrow + s + 16);
            return f;
        };

        v8f c = {};
        PVFrag pf0 = pvload(0);
        for (int s = 0; s < S_LEN; s += 64) {
            PVFrag pf1 = pvload(s + 32);
            c = WMMA_F16(combine8(pf0.p0, pf0.p1), combine8(pf0.u0, pf0.u1), c);
            pf0 = pvload((s + 64 < S_LEN) ? s + 64 : 0);
            c = WMMA_F16(combine8(pf1.p0, pf1.p1), combine8(pf1.u0, pf1.u1), c);
        }
#pragma unroll
        for (int v = 0; v < 8; ++v) {
            int t = t0 + mh * 16 + v + 8 * hi;
            int d = nt * 16 + lm;
            attn_out[((size_t)t * B_SZ + b) * E_DIM + h * HD + d] = (_Float16)c[v];
        }
    }
}

// ---------------------------------------------------------------- launcher
extern "C" void kernel_launch(void* const* d_in, const int* in_sizes, int n_in,
                              void* d_out, int out_size, void* d_ws, size_t ws_size,
                              hipStream_t stream) {
    (void)in_sizes; (void)n_in; (void)out_size; (void)ws_size;

    const float* query = (const float*)d_in[0];
    const float* key_  = (const float*)d_in[1];
    const float* value = (const float*)d_in[2];
    const float* qw  = (const float*)d_in[3];  const float* qb  = (const float*)d_in[4];
    const float* qla = (const float*)d_in[5];  const float* qlb = (const float*)d_in[6];
    const float* kw  = (const float*)d_in[7];  const float* kbi = (const float*)d_in[8];
    const float* kla = (const float*)d_in[9];  const float* klb = (const float*)d_in[10];
    const float* vw  = (const float*)d_in[11]; const float* vb  = (const float*)d_in[12];
    const float* vla = (const float*)d_in[13]; const float* vlb = (const float*)d_in[14];
    const float* ow  = (const float*)d_in[15]; const float* ob  = (const float*)d_in[16];
    const float* ola = (const float*)d_in[17]; const float* olb = (const float*)d_in[18];

    float* out_f  = (float*)d_out;                                   // (T,B,E)
    float* attn_w = out_f + (size_t)T_LEN * B_SZ * E_DIM;            // (B,T,S)

    char* ws = (char*)d_ws;
    const size_t SZ_X   = (size_t)N_ROWS * E_DIM * sizeof(_Float16);             // 8 MB
    const size_t SZ_W   = (size_t)E_DIM * E_DIM * sizeof(_Float16);              // 2 MB
    const size_t SZ_QKV = (size_t)B_SZ * H_NUM * S_LEN * HD * sizeof(_Float16);  // 8 MB
    _Float16* xq  = (_Float16*)ws;  ws += SZ_X;
    _Float16* xk  = (_Float16*)ws;  ws += SZ_X;
    _Float16* xv  = (_Float16*)ws;  ws += SZ_X;
    _Float16* wqh = (_Float16*)ws;  ws += SZ_W;
    _Float16* wkh = (_Float16*)ws;  ws += SZ_W;
    _Float16* wvh = (_Float16*)ws;  ws += SZ_W;
    _Float16* woh = (_Float16*)ws;  ws += SZ_W;
    _Float16* qhd = (_Float16*)ws;  ws += SZ_QKV;
    _Float16* khd = (_Float16*)ws;  ws += SZ_QKV;
    _Float16* vTd = (_Float16*)ws;  ws += SZ_QKV;
    _Float16* attn_out = xq;  // xq dead after q projection; reuse (stream-ordered)

    const int nX = N_ROWS * E_DIM;                 // 4,194,304
    cvt_f32_to_f16<<<4096, 256, 0, stream>>>(query, xq, nX);
    cvt_f32_to_f16<<<4096, 256, 0, stream>>>(key_,  xk, nX);
    cvt_f32_to_f16<<<4096, 256, 0, stream>>>(value, xv, nX);

    build_weff<<<E_DIM * E_DIM / 256, 256, 0, stream>>>(qw, qla, qlb, wqh);
    build_weff<<<E_DIM * E_DIM / 256, 256, 0, stream>>>(kw, kla, klb, wkh);
    build_weff<<<E_DIM * E_DIM / 256, 256, 0, stream>>>(vw, vla, vlb, wvh);
    build_weff<<<E_DIM * E_DIM / 256, 256, 0, stream>>>(ow, ola, olb, woh);

    zero_f32<<<4096, 256, 0, stream>>>(attn_w, B_SZ * T_LEN * S_LEN);

    dim3 ggrid(N_ROWS / 256, E_DIM / 64);          // (16, 16)
    gemm_proj<<<ggrid, 256, 0, stream>>>(xq, wqh, qb,  (void*)qhd, 0, QK_SCALE);
    gemm_proj<<<ggrid, 256, 0, stream>>>(xk, wkh, kbi, (void*)khd, 0, 1.0f);
    gemm_proj<<<ggrid, 256, 0, stream>>>(xv, wvh, vb,  (void*)vTd, 1, 1.0f);

    dim3 agrid(T_LEN / 32, B_SZ * H_NUM);          // (64, 32)
    size_t smem = (size_t)32 * SPITCH * sizeof(_Float16);   // 131,584 B
    attn_kernel<<<agrid, 256, smem, stream>>>(qhd, khd, vTd, attn_out, attn_w);

    gemm_proj<<<ggrid, 256, 0, stream>>>(attn_out, woh, ob, (void*)out_f, 2, 1.0f);
}
